// GCU_H_60241211294346
// MI455X (gfx1250) — matrix operand order repaired and checked by reading, not compiled, not run on hardware
//
#include <hip/hip_runtime.h>

// ---------------------------------------------------------------------------
// Fused height-axis attention for MI455X (gfx1250), wave32 + WMMA bf16.
// One workgroup (256 thr = 8 wave32) per (b, w) column. Everything in LDS.
// ---------------------------------------------------------------------------

typedef __attribute__((ext_vector_type(16))) __bf16 v16bf;
typedef __attribute__((ext_vector_type(8)))  __bf16 v8bf;
typedef __attribute__((ext_vector_type(8)))  float  v8f;

#define BB   16
#define CC   256
#define HH   128
#define WW   128
#define C4D  64
#define HWsz (HH * WW)

// ---- LDS layout (bytes); rows padded for bank-conflict-free access --------
#define XROW  264   // X staging: [128][264] u16 (bf16), 67584 B
#define EROW  132   // energy:    [128][132] f32, 67584 B (aliases X)
#define QKROW 72    // q/k:       [128][72]  u16, 18432 B each
#define AROW  136   // att:       [128][136] u16, 34816 B (aliases q+k)
#define VROW  136   // v:         [256][136] u16, 69632 B
#define X_OFF   0
#define Q_OFF   67584
#define K_OFF   86016
#define ATT_OFF 67584
#define V_OFF   104448
#define SMEM_BYTES 174080

__device__ __forceinline__ unsigned short f2bf_bits(float x) {
  unsigned int u = __float_as_uint(x);
  unsigned int r = u + 0x7fffu + ((u >> 16) & 1u);   // round-to-nearest-even
  return (unsigned short)(r >> 16);
}

__device__ __forceinline__ v8f wmma_bf16(v16bf a, v16bf b, v8f c) {
  // v_wmma_f32_16x16x32_bf16: (neg_a, A, neg_b, B, c_mod, C, reuse_a, reuse_b)
  return __builtin_amdgcn_wmma_f32_16x16x32_bf16(false, a, false, b,
                                                 (short)0, c, false, false);
}

// A fragment (16x32, bf16), source row-major [row][K], rowStride in elements.
// Layout: lanes 0-15 hold K {0..15}; lanes 16-31 hold K {16..31} (matches load_b).
__device__ __forceinline__ v16bf load_a(const unsigned short* sm, int row0,
                                        int rowStride, int k0, int lane) {
  const int r  = row0 + (lane & 15);
  const int ks = (lane >> 4) * 16;
  v8bf lo = *(const v8bf*)(sm + r * rowStride + k0 + ks);
  v8bf hi = *(const v8bf*)(sm + r * rowStride + k0 + ks + 8);
  v16bf a;
#pragma unroll
  for (int i = 0; i < 8; ++i) { a[i] = lo[i]; a[i + 8] = hi[i]; }
  return a;
}

// B fragment (32x16, bf16) from source stored as [N][K] row-major:
// lane = column N, reads 16 contiguous K at (lane>>4)*16.
__device__ __forceinline__ v16bf load_b(const unsigned short* sm, int col0,
                                        int rowStride, int k0, int lane) {
  const int c  = col0 + (lane & 15);
  const int ks = k0 + (lane >> 4) * 16;
  return *(const v16bf*)(sm + c * rowStride + ks);
}

// Stage one input column src[b, :, :, w] -> xs[h][c] as bf16.
__device__ __forceinline__ void stage_x(const float* __restrict__ src,
                                        unsigned short* xs, int b, int w,
                                        int tid) {
  const size_t base = (size_t)b * CC * HWsz + w;
  for (int idx = tid; idx < CC * HH; idx += 256) {
    const int c = idx >> 7;      // channel
    const int h = idx & 127;     // height
    float v = src[base + (size_t)c * HWsz + (size_t)h * WW];
    xs[h * XROW + c] = f2bf_bits(v);
  }
}

// Projection GEMM: Out[h, oc] = sum_c W[oc,c] * X[h,c] + bias[oc]
//   A = X^T tiles from LDS, B = bf16 weights [oc][256] from global.
//   TRANS=false: dst[h*stride + oc] (q/k layout [h][d])
//   TRANS=true : dst[oc*stride + h] (v layout [c][g])
template <int NT, bool TRANS>
__device__ __forceinline__ void proj_gemm(const unsigned short* __restrict__ xs,
                                          const unsigned short* __restrict__ wbf,
                                          const float* __restrict__ bias,
                                          unsigned short* dst, int dstStride,
                                          int lane, int wave) {
  for (int nt = wave; nt < NT; nt += 8) {
    const int oc = nt * 16 + (lane & 15);
    const float bv = bias[oc];
    v8f acc[8];
#pragma unroll
    for (int m = 0; m < 8; ++m)
#pragma unroll
      for (int r = 0; r < 8; ++r) acc[m][r] = 0.0f;

    for (int k0 = 0; k0 < CC; k0 += 32) {
      v16bf bt = load_b(wbf, nt * 16, CC, k0, lane);
#pragma unroll
      for (int m = 0; m < 8; ++m) {
        v16bf at = load_a(xs, m * 16, XROW, k0, lane);
        acc[m] = wmma_bf16(at, bt, acc[m]);
      }
    }
#pragma unroll
    for (int m = 0; m < 8; ++m) {
#pragma unroll
      for (int r = 0; r < 8; ++r) {
        const int h = m * 16 + r + 8 * (lane >> 4);
        const unsigned short bv16 = f2bf_bits(acc[m][r] + bv);
        if (TRANS) dst[oc * dstStride + h] = bv16;
        else       dst[h * dstStride + oc] = bv16;
      }
    }
  }
}

__global__ void __launch_bounds__(256, 1)
GCU_attn_fused_kernel(const float* __restrict__ flow,
                      const float* __restrict__ de_out,
                      const unsigned short* __restrict__ wbf,
                      const float* __restrict__ bq,
                      const float* __restrict__ bk,
                      const float* __restrict__ bv,
                      float* __restrict__ out) {
  extern __shared__ char smem[];
  unsigned short* xs    = (unsigned short*)(smem + X_OFF);
  unsigned short* qsm   = (unsigned short*)(smem + Q_OFF);
  unsigned short* ksm   = (unsigned short*)(smem + K_OFF);
  unsigned short* vsm   = (unsigned short*)(smem + V_OFF);
  unsigned short* attsm = (unsigned short*)(smem + ATT_OFF);
  float*          esm   = (float*)(smem + X_OFF);

  const int tid  = threadIdx.x;
  const int lane = tid & 31;
  const int wave = tid >> 5;
  const int w = blockIdx.x;
  const int b = blockIdx.y;

  const unsigned short* wq_bf = wbf;
  const unsigned short* wk_bf = wbf + 64 * CC;
  const unsigned short* wv_bf = wbf + 128 * CC;

  // ---- Phase 1: stage flow column --------------------------------------
  stage_x(flow, xs, b, w, tid);
  __syncthreads();

  // ---- Phase 2: k = Wk@Xf + bk -> ksm[g][d];  v = Wv@Xf + bv -> vsm[c][g]
  proj_gemm<4, false>(xs, wk_bf, bk, ksm, QKROW, lane, wave);
  proj_gemm<16, true>(xs, wv_bf, bv, vsm, VROW, lane, wave);
  __syncthreads();

  // ---- Phase 3: stage de_out column (overwrites X buffer) --------------
  stage_x(de_out, xs, b, w, tid);
  __syncthreads();

  // ---- Phase 4: q = Wq@Xd + bq -> qsm[h][d] ----------------------------
  proj_gemm<4, false>(xs, wq_bf, bq, qsm, QKROW, lane, wave);
  __syncthreads();

  // ---- Phase 5: energy[h][g] = sum_d q[h][d]*k[g][d] (K=64) ------------
  for (int t = wave; t < 64; t += 8) {
    const int mt = t >> 3, nt = t & 7;
    v8f acc;
#pragma unroll
    for (int r = 0; r < 8; ++r) acc[r] = 0.0f;
#pragma unroll
    for (int k0 = 0; k0 < C4D; k0 += 32) {
      v16bf at = load_a(qsm, mt * 16, QKROW, k0, lane);
      v16bf bt = load_b(ksm, nt * 16, QKROW, k0, lane);
      acc = wmma_bf16(at, bt, acc);
    }
#pragma unroll
    for (int r = 0; r < 8; ++r) {
      const int h = mt * 16 + r + 8 * (lane >> 4);
      const int g = nt * 16 + (lane & 15);
      esm[h * EROW + g] = acc[r];
    }
  }
  __syncthreads();

  // ---- Phase 6: softmax over g, row per thread; att -> bf16 ------------
  if (tid < HH) {
    const int h = tid;
    float mx = -3.0e38f;
    for (int g = 0; g < HH; ++g) mx = fmaxf(mx, esm[h * EROW + g]);
    float s = 0.0f;
    for (int g = 0; g < HH; ++g) {
      float e = __expf(esm[h * EROW + g] - mx);
      esm[h * EROW + g] = e;
      s += e;
    }
    const float inv = 1.0f / s;
    for (int g = 0; g < HH; ++g)
      attsm[h * AROW + g] = f2bf_bits(esm[h * EROW + g] * inv);
  }
  __syncthreads();

  // ---- Phase 7: out[c][h] = sum_g v[c][g]*att[h][g] (K=128) ------------
  const size_t obase = (size_t)b * CC * HWsz + w;
  for (int t = wave; t < 128; t += 8) {
    const int mt = t >> 4;           // h tile (8)
    const int nt = t & 15;           // c tile (16)
    v8f acc;
#pragma unroll
    for (int r = 0; r < 8; ++r) acc[r] = 0.0f;
#pragma unroll
    for (int k0 = 0; k0 < HH; k0 += 32) {
      v16bf at = load_a(attsm, mt * 16, AROW, k0, lane);
      v16bf bt = load_b(vsm, nt * 16, VROW, k0, lane);
      acc = wmma_bf16(at, bt, acc);
    }
    const int c = nt * 16 + (lane & 15);
#pragma unroll
    for (int r = 0; r < 8; ++r) {
      const int h = mt * 16 + r + 8 * (lane >> 4);
      out[obase + (size_t)c * HWsz + (size_t)h * WW] = acc[r];
    }
  }
}

// Convert f32 weights -> bf16 once into d_ws: [Wq 64x256][Wk 64x256][Wv 256x256]
__global__ void GCU_cvt_w_kernel(const float* __restrict__ Wq,
                                 const float* __restrict__ Wk,
                                 const float* __restrict__ Wv,
                                 unsigned short* __restrict__ wbf) {
  const int idx = blockIdx.x * 256 + threadIdx.x;
  float v;
  if (idx < 16384)       v = Wq[idx];
  else if (idx < 32768)  v = Wk[idx - 16384];
  else if (idx < 98304)  v = Wv[idx - 32768];
  else return;
  wbf[idx] = f2bf_bits(v);
}

extern "C" void kernel_launch(void* const* d_in, const int* in_sizes, int n_in,
                              void* d_out, int out_size, void* d_ws,
                              size_t ws_size, hipStream_t stream) {
  const float* flow   = (const float*)d_in[0];
  const float* de_out = (const float*)d_in[1];
  const float* Wq     = (const float*)d_in[2];
  const float* bq     = (const float*)d_in[3];
  const float* Wk     = (const float*)d_in[4];
  const float* bk     = (const float*)d_in[5];
  const float* Wv     = (const float*)d_in[6];
  const float* bv     = (const float*)d_in[7];
  float* out = (float*)d_out;
  unsigned short* wbf = (unsigned short*)d_ws;   // 192 KB of workspace used

  hipLaunchKernelGGL(GCU_cvt_w_kernel, dim3(384), dim3(256), 0, stream,
                     Wq, Wk, Wv, wbf);
  hipLaunchKernelGGL(GCU_attn_fused_kernel, dim3(WW, BB), dim3(256),
                     SMEM_BYTES, stream, flow, de_out, wbf, bq, bk, bv, out);
}
